// CustomSeq2SeqModel_73650099192450
// MI455X (gfx1250) — matrix operand order repaired and checked by reading, not compiled, not run on hardware
//
#include <hip/hip_runtime.h>

#define B    32
#define S    128
#define E    256
#define H    256
#define HD   512
#define VOUT 16000
#define TLEN 128

typedef __attribute__((ext_vector_type(16))) __bf16 v16bf;
typedef __attribute__((ext_vector_type(8)))  float  v8f;

// ---------------- WMMA helpers (bf16 16x16x32, fp32 accum) ----------------

__device__ __forceinline__ v8f wmma_bf16(v16bf a, v16bf b, v8f c) {
  return __builtin_amdgcn_wmma_f32_16x16x32_bf16(
      /*neg_a=*/false, a, /*neg_b=*/false, b,
      /*c_mod=*/(short)0, c, /*reuse_a=*/false, /*reuse_b=*/false);
}

// A fragment: row-major A, one row per lane-group-of-16; per ISA layout,
// lane (l16, half): elems 0..7 -> K = k0 + half*8 + e ; elems 8..15 -> K = k0+16+half*8+e
__device__ __forceinline__ v16bf load_a_row(const __bf16* rowptr, int k0, int half) {
  v16bf a;
  const __bf16* p = rowptr + k0 + half * 8;
#pragma unroll
  for (int e = 0; e < 8; ++e) { a[e] = p[e]; a[e + 8] = p[16 + e]; }
  return a;
}

// B fragment: W is row-major [N, K]; lane (l16, half) reads row n=col:
// elems 0..7 -> K = k0 + half*8 + e ; elems 8..15 -> K = k0+16+half*8+e
__device__ __forceinline__ v16bf load_b_row(const __bf16* rowptr, int k0, int half) {
  v16bf b;
  const __bf16* p = rowptr + k0 + half * 8;
#pragma unroll
  for (int e = 0; e < 8; ++e) { b[e] = p[e]; b[e + 8] = p[16 + e]; }
  return b;
}

__device__ __forceinline__ float sigmoidf_(float x) { return 1.f / (1.f + __expf(-x)); }

__device__ __forceinline__ unsigned long long make_key(float v, unsigned col) {
  unsigned u = __float_as_uint(v);
  u = (u & 0x80000000u) ? ~u : (u | 0x80000000u);          // monotone order-preserving map
  return ((unsigned long long)u << 32) |
         (unsigned long long)(0xFFFFFFFFu - col);          // prefer smaller col on ties
}

// ---------------- CDNA5 async global->LDS staging ----------------
// global_load_async_to_lds_b128 vDstLdsOff, v[addr], off   (tracked with ASYNCcnt)

__device__ __forceinline__ void async_ld_b128(unsigned ldsByteOff, const void* gaddr) {
  asm volatile("global_load_async_to_lds_b128 %0, %1, off"
               :: "v"(ldsByteOff), "v"(gaddr)
               : "memory");
}

__device__ __forceinline__ void wait_async0() {
#if __has_builtin(__builtin_amdgcn_s_wait_asynccnt)
  __builtin_amdgcn_s_wait_asynccnt(0);
#else
  asm volatile("s_wait_asynccnt 0" ::: "memory");
#endif
}

// generic-pointer low 32 bits of a __shared__ object == LDS byte offset on amdgcn
__device__ __forceinline__ unsigned lds_off(const void* p) {
  return (unsigned)(unsigned long long)p;
}

// ---------------- prep kernels ----------------

extern "C" __global__ void k_cast(const float* __restrict__ src, __bf16* __restrict__ dst, int n) {
  for (int i = blockIdx.x * blockDim.x + threadIdx.x; i < n; i += gridDim.x * blockDim.x)
    dst[i] = (__bf16)src[i];
}

extern "C" __global__ void k_embed(const int* __restrict__ inputs, const float* __restrict__ emb,
                                   __bf16* __restrict__ xs, int total) {
  for (int i = blockIdx.x * blockDim.x + threadIdx.x; i < total; i += gridDim.x * blockDim.x) {
    int e = i % E; int r = i / E; int b = r % B; int s = r / B;
    xs[i] = (__bf16)emb[(size_t)inputs[b * S + s] * E + e];
  }
}

extern "C" __global__ void k_first_row(float* __restrict__ out, const int* __restrict__ cls) {
  int i = blockIdx.x * blockDim.x + threadIdx.x;
  if (i >= B * VOUT) return;
  int b = i / VOUT, v = i % VOUT;
  out[(size_t)b * TLEN * VOUT + v] = (v == cls[0]) ? 1.f : 0.f;
}

extern "C" __global__ void k_init_tok(int* tok, const int* cls) {
  int i = threadIdx.x; if (i < B) tok[i] = cls[0];
}
extern "C" __global__ void k_clear_keys(unsigned long long* keys) {
  int i = threadIdx.x; if (i < B) keys[i] = 0ull;
}
extern "C" __global__ void k_next_tok(const unsigned long long* keys, int* tok) {
  int i = threadIdx.x;
  if (i < B) tok[i] = (int)(0xFFFFFFFFu - (unsigned)(keys[i] & 0xFFFFFFFFull));
}

// ---------------- encoder: persistent scan, one workgroup per direction ----------------
// Block = 512 threads = 16 waves (2x VGPR budget vs 32 waves -> no spills).
// Wave w owns hidden slice ng=w (16 cols) and BOTH batch tiles: each B-fragment
// feeds 2 WMMAs. i/f/g/o x {m0,m1} = 8 accumulators; cell update in registers.

extern "C" __global__ void __launch_bounds__(512)
k_enc_scan(const __bf16* __restrict__ xin, int Kin, __bf16* __restrict__ yout,
           const __bf16* __restrict__ Wih_f, const __bf16* __restrict__ Whh_f, const float* __restrict__ b_f,
           const __bf16* __restrict__ Wih_b, const __bf16* __restrict__ Whh_b, const float* __restrict__ b_b,
           float* __restrict__ hfin, float* __restrict__ cfin) {
  __shared__ __attribute__((aligned(16))) __bf16 hL[B * H];
  __shared__ float  cL[B * H];
  const int tid = threadIdx.x;
  const int dir = blockIdx.x;
  const __bf16* Wih = dir ? Wih_b : Wih_f;
  const __bf16* Whh = dir ? Whh_b : Whh_f;
  const float*  bia = dir ? b_b : b_f;

  for (int i = tid; i < B * H; i += 512) { hL[i] = (__bf16)0.f; cL[i] = 0.f; }
  __syncthreads();

  const int lane = tid & 31, wv = tid >> 5;   // wv = 0..15
  const int half = lane >> 4, l16 = lane & 15;
  const int ng = wv;
  const int j = ng * 16 + l16;                // hidden column owned by this lane
  const int arow0 = l16, arow1 = 16 + l16;    // batch rows for the two A fragments

  const __bf16* wih0 = Wih + (size_t)(0 * H + j) * Kin;
  const __bf16* wih1 = Wih + (size_t)(1 * H + j) * Kin;
  const __bf16* wih2 = Wih + (size_t)(2 * H + j) * Kin;
  const __bf16* wih3 = Wih + (size_t)(3 * H + j) * Kin;
  const __bf16* whh0 = Whh + (size_t)(0 * H + j) * H;
  const __bf16* whh1 = Whh + (size_t)(1 * H + j) * H;
  const __bf16* whh2 = Whh + (size_t)(2 * H + j) * H;
  const __bf16* whh3 = Whh + (size_t)(3 * H + j) * H;

  for (int s = 0; s < S; ++s) {
    const int p = dir ? (S - 1 - s) : s;
    const __bf16* xrow0 = xin + ((size_t)p * B + arow0) * Kin;
    const __bf16* xrow1 = xin + ((size_t)p * B + arow1) * Kin;
    v8f acc[8] = {};                          // acc[g*2 + m]

    // gates += x @ Wih^T : 2 A-loads + 4 B-loads -> 8 WMMAs
    for (int k0 = 0; k0 < Kin; k0 += 32) {
      v16bf a0 = load_a_row(xrow0, k0, half);
      v16bf a1 = load_a_row(xrow1, k0, half);
      v16bf b0 = load_b_row(wih0, k0, half);
      v16bf b1 = load_b_row(wih1, k0, half);
      v16bf b2 = load_b_row(wih2, k0, half);
      v16bf b3 = load_b_row(wih3, k0, half);
      acc[0] = wmma_bf16(a0, b0, acc[0]);  acc[1] = wmma_bf16(a1, b0, acc[1]);
      acc[2] = wmma_bf16(a0, b1, acc[2]);  acc[3] = wmma_bf16(a1, b1, acc[3]);
      acc[4] = wmma_bf16(a0, b2, acc[4]);  acc[5] = wmma_bf16(a1, b2, acc[5]);
      acc[6] = wmma_bf16(a0, b3, acc[6]);  acc[7] = wmma_bf16(a1, b3, acc[7]);
    }
    // gates += h @ Whh^T   (h lives in LDS)
    for (int k0 = 0; k0 < H; k0 += 32) {
      v16bf a0 = load_a_row(&hL[arow0 * H], k0, half);
      v16bf a1 = load_a_row(&hL[arow1 * H], k0, half);
      v16bf b0 = load_b_row(whh0, k0, half);
      v16bf b1 = load_b_row(whh1, k0, half);
      v16bf b2 = load_b_row(whh2, k0, half);
      v16bf b3 = load_b_row(whh3, k0, half);
      acc[0] = wmma_bf16(a0, b0, acc[0]);  acc[1] = wmma_bf16(a1, b0, acc[1]);
      acc[2] = wmma_bf16(a0, b1, acc[2]);  acc[3] = wmma_bf16(a1, b1, acc[3]);
      acc[4] = wmma_bf16(a0, b2, acc[4]);  acc[5] = wmma_bf16(a1, b2, acc[5]);
      acc[6] = wmma_bf16(a0, b3, acc[6]);  acc[7] = wmma_bf16(a1, b3, acc[7]);
    }
    __syncthreads();   // everyone done reading hL

    const float bi = bia[0 * H + j], bff = bia[1 * H + j];
    const float bg = bia[2 * H + j], bo = bia[3 * H + j];
#pragma unroll
    for (int m = 0; m < 2; ++m) {
#pragma unroll
      for (int v = 0; v < 8; ++v) {
        const int brow = m * 16 + half * 8 + v;   // C/D: lanes 0-15 -> M=v, 16-31 -> M=8+v
        float iv = sigmoidf_(acc[0 * 2 + m][v] + bi);
        float fv = sigmoidf_(acc[1 * 2 + m][v] + bff);
        float gv = tanhf(acc[2 * 2 + m][v] + bg);
        float ov = sigmoidf_(acc[3 * 2 + m][v] + bo);
        float cv = fv * cL[brow * H + j] + iv * gv;
        float hv = ov * tanhf(cv);
        cL[brow * H + j] = cv;
        hL[brow * H + j] = (__bf16)hv;
        if (yout) yout[((size_t)p * B + brow) * (2 * H) + dir * H + j] = (__bf16)hv;
      }
    }
    __syncthreads();   // hL updated before next step reads it
  }

  // export final h/c into the decoder state slices: [b, dir*H + j]
  for (int i = tid; i < B * H; i += 512) {
    int b_ = i / H, jj = i % H;
    hfin[b_ * HD + dir * H + jj] = (float)hL[i];
    cfin[b_ * HD + dir * H + jj] = cL[i];
  }
}

// ---------------- decoder LSTM cell (Hd=512) ----------------
// 32 wave-tasks (ng = 0..31), each wave computes both batch tiles -> each weight
// fragment feeds 2 WMMAs. h_prev (32 KB) async-staged into LDS once per block.

extern "C" __global__ void __launch_bounds__(256)
k_dec_cell(const __bf16* __restrict__ xsrc, int xIsEmb, const int* __restrict__ tok,
           const __bf16* __restrict__ oemb, int Kin,
           const __bf16* __restrict__ hprev, __bf16* __restrict__ hnext, float* __restrict__ c,
           const __bf16* __restrict__ Wih, const __bf16* __restrict__ Whh,
           const float* __restrict__ bias) {
  __shared__ __attribute__((aligned(16))) __bf16 hS[B * HD];   // 32 KB
  const int tid = threadIdx.x;

  // async-stage h_prev: 32*512*2 B = 2048 x b128 ; 256 threads -> 8 each
  {
    const unsigned base = lds_off(&hS[0]);
    const char* g = (const char*)hprev;
#pragma unroll
    for (int i = 0; i < 8; ++i) {
      const unsigned off = (unsigned)((tid + i * 256) * 16);
      async_ld_b128(base + off, g + off);
    }
    wait_async0();
  }
  __syncthreads();

  const int lane = tid & 31, wv = tid >> 5;
  const int ng = blockIdx.x * 8 + wv;          // 0..31
  const int half = lane >> 4, l16 = lane & 15;
  const int j = ng * 16 + l16;
  const int arow0 = l16, arow1 = 16 + l16;

  const __bf16* xrow0 = xIsEmb ? (oemb + (size_t)tok[arow0] * E)
                               : (xsrc + (size_t)arow0 * Kin);
  const __bf16* xrow1 = xIsEmb ? (oemb + (size_t)tok[arow1] * E)
                               : (xsrc + (size_t)arow1 * Kin);
  const __bf16* wih0 = Wih + (size_t)(0 * HD + j) * Kin;
  const __bf16* wih1 = Wih + (size_t)(1 * HD + j) * Kin;
  const __bf16* wih2 = Wih + (size_t)(2 * HD + j) * Kin;
  const __bf16* wih3 = Wih + (size_t)(3 * HD + j) * Kin;
  const __bf16* whh0 = Whh + (size_t)(0 * HD + j) * HD;
  const __bf16* whh1 = Whh + (size_t)(1 * HD + j) * HD;
  const __bf16* whh2 = Whh + (size_t)(2 * HD + j) * HD;
  const __bf16* whh3 = Whh + (size_t)(3 * HD + j) * HD;

  v8f acc[8] = {};                             // acc[g*2 + m]
  for (int k0 = 0; k0 < Kin; k0 += 32) {
    v16bf a0 = load_a_row(xrow0, k0, half);
    v16bf a1 = load_a_row(xrow1, k0, half);
    v16bf b0 = load_b_row(wih0, k0, half);
    v16bf b1 = load_b_row(wih1, k0, half);
    v16bf b2 = load_b_row(wih2, k0, half);
    v16bf b3 = load_b_row(wih3, k0, half);
    acc[0] = wmma_bf16(a0, b0, acc[0]);  acc[1] = wmma_bf16(a1, b0, acc[1]);
    acc[2] = wmma_bf16(a0, b1, acc[2]);  acc[3] = wmma_bf16(a1, b1, acc[3]);
    acc[4] = wmma_bf16(a0, b2, acc[4]);  acc[5] = wmma_bf16(a1, b2, acc[5]);
    acc[6] = wmma_bf16(a0, b3, acc[6]);  acc[7] = wmma_bf16(a1, b3, acc[7]);
  }
  for (int k0 = 0; k0 < HD; k0 += 32) {
    v16bf a0 = load_a_row(&hS[arow0 * HD], k0, half);
    v16bf a1 = load_a_row(&hS[arow1 * HD], k0, half);
    v16bf b0 = load_b_row(whh0, k0, half);
    v16bf b1 = load_b_row(whh1, k0, half);
    v16bf b2 = load_b_row(whh2, k0, half);
    v16bf b3 = load_b_row(whh3, k0, half);
    acc[0] = wmma_bf16(a0, b0, acc[0]);  acc[1] = wmma_bf16(a1, b0, acc[1]);
    acc[2] = wmma_bf16(a0, b1, acc[2]);  acc[3] = wmma_bf16(a1, b1, acc[3]);
    acc[4] = wmma_bf16(a0, b2, acc[4]);  acc[5] = wmma_bf16(a1, b2, acc[5]);
    acc[6] = wmma_bf16(a0, b3, acc[6]);  acc[7] = wmma_bf16(a1, b3, acc[7]);
  }

  const float bi = bias[0 * HD + j], bff = bias[1 * HD + j];
  const float bg = bias[2 * HD + j], bo = bias[3 * HD + j];
#pragma unroll
  for (int m = 0; m < 2; ++m) {
#pragma unroll
    for (int v = 0; v < 8; ++v) {
      const int brow = m * 16 + half * 8 + v;
      float iv = sigmoidf_(acc[0 * 2 + m][v] + bi);
      float fv = sigmoidf_(acc[1 * 2 + m][v] + bff);
      float gv = tanhf(acc[2 * 2 + m][v] + bg);
      float ov = sigmoidf_(acc[3 * 2 + m][v] + bo);
      float cv = fv * c[brow * HD + j] + iv * gv;
      float hv = ov * tanhf(cv);
      c[brow * HD + j] = cv;
      hnext[brow * HD + j] = (__bf16)hv;
    }
  }
}

// ---------------- decoder projection + fused argmax ----------------
// [32,512] x [512,16000]; 250 wave-tasks, each = 4 N-tiles x 2 M-tiles.
// Each weight fragment feeds 2 WMMAs; h1 async-staged in LDS.

extern "C" __global__ void __launch_bounds__(256)
k_dec_proj(const __bf16* __restrict__ h1, const __bf16* __restrict__ W,
           const float* __restrict__ bias, float* __restrict__ out, int t,
           unsigned long long* __restrict__ keys) {
  __shared__ __attribute__((aligned(16))) __bf16 hS[B * HD];   // 32 KB
  const int tid = threadIdx.x;

  // async-stage h1: 2048 x b128 ; 256 threads -> 8 each
  {
    const unsigned base = lds_off(&hS[0]);
    const char* g = (const char*)h1;
#pragma unroll
    for (int i = 0; i < 8; ++i) {
      const unsigned off = (unsigned)((tid + i * 256) * 16);
      async_ld_b128(base + off, g + off);
    }
    wait_async0();
  }
  __syncthreads();                             // all waves reach this before any compute

  const int lane = tid & 31, wv = tid >> 5;
  const int task = blockIdx.x * 8 + wv;        // 0..255 ; valid tasks 0..249
  if (task >= 250) return;
  const int ntg = task;                        // group of 4 N-tiles
  const int half = lane >> 4, l16 = lane & 15;
  const __bf16* hrow0 = &hS[(0 * 16 + l16) * HD];
  const __bf16* hrow1 = &hS[(1 * 16 + l16) * HD];

  const __bf16* wr0 = W + (size_t)(ntg * 64 + 0 * 16 + l16) * HD;
  const __bf16* wr1 = W + (size_t)(ntg * 64 + 1 * 16 + l16) * HD;
  const __bf16* wr2 = W + (size_t)(ntg * 64 + 2 * 16 + l16) * HD;
  const __bf16* wr3 = W + (size_t)(ntg * 64 + 3 * 16 + l16) * HD;

  v8f acc[8] = {};                             // acc[i*2 + m]
  for (int k0 = 0; k0 < HD; k0 += 32) {
    __builtin_prefetch(wr0 + k0 + 64, 0, 1);   // global_prefetch_b8 on the streaming weights
    v16bf a0 = load_a_row(hrow0, k0, half);
    v16bf a1 = load_a_row(hrow1, k0, half);
    v16bf b0 = load_b_row(wr0, k0, half);
    v16bf b1 = load_b_row(wr1, k0, half);
    v16bf b2 = load_b_row(wr2, k0, half);
    v16bf b3 = load_b_row(wr3, k0, half);
    acc[0] = wmma_bf16(a0, b0, acc[0]);  acc[1] = wmma_bf16(a1, b0, acc[1]);
    acc[2] = wmma_bf16(a0, b1, acc[2]);  acc[3] = wmma_bf16(a1, b1, acc[3]);
    acc[4] = wmma_bf16(a0, b2, acc[4]);  acc[5] = wmma_bf16(a1, b2, acc[5]);
    acc[6] = wmma_bf16(a0, b3, acc[6]);  acc[7] = wmma_bf16(a1, b3, acc[7]);
  }

#pragma unroll
  for (int i = 0; i < 4; ++i) {
    const int col = ntg * 64 + i * 16 + l16;
    const float bb = bias[col];
#pragma unroll
    for (int m = 0; m < 2; ++m) {
#pragma unroll
      for (int v = 0; v < 8; ++v) {
        const int brow = m * 16 + half * 8 + v;
        float val = acc[i * 2 + m][v] + bb;
        out[((size_t)brow * TLEN + t) * VOUT + col] = val;

        // per-row argmax across the 16 columns of this tile (stays within each half-wave)
        float mv = val; unsigned mc = (unsigned)col;
#pragma unroll
        for (int msk = 8; msk >= 1; msk >>= 1) {
          float ov = __shfl_xor(mv, msk, 32);
          unsigned oc = (unsigned)__shfl_xor((int)mc, msk, 32);
          if (ov > mv || (ov == mv && oc < mc)) { mv = ov; mc = oc; }
        }
        if (l16 == 0) atomicMax(&keys[brow], make_key(mv, mc));
      }
    }
  }
}

// ---------------- host orchestration ----------------

extern "C" void kernel_launch(void* const* d_in, const int* in_sizes, int n_in,
                              void* d_out, int out_size, void* d_ws, size_t ws_size,
                              hipStream_t stream) {
  (void)in_sizes; (void)n_in; (void)out_size; (void)ws_size;
  const int*   inputs = (const int*)d_in[0];
  const int*   cls    = (const int*)d_in[1];
  const float* in_emb = (const float*)d_in[2];
  const float* o_emb  = (const float*)d_in[3];
  const float* e0f_ih = (const float*)d_in[4];
  const float* e0f_hh = (const float*)d_in[5];
  const float* e0f_b  = (const float*)d_in[6];
  const float* e0b_ih = (const float*)d_in[7];
  const float* e0b_hh = (const float*)d_in[8];
  const float* e0b_b  = (const float*)d_in[9];
  const float* e1f_ih = (const float*)d_in[10];
  const float* e1f_hh = (const float*)d_in[11];
  const float* e1f_b  = (const float*)d_in[12];
  const float* e1b_ih = (const float*)d_in[13];
  const float* e1b_hh = (const float*)d_in[14];
  const float* e1b_b  = (const float*)d_in[15];
  const float* d0_ih  = (const float*)d_in[16];
  const float* d0_hh  = (const float*)d_in[17];
  const float* d0_b   = (const float*)d_in[18];
  const float* d1_ih  = (const float*)d_in[19];
  const float* d1_hh  = (const float*)d_in[20];
  const float* d1_b   = (const float*)d_in[21];
  const float* linW   = (const float*)d_in[22];
  const float* linb   = (const float*)d_in[23];
  float* out = (float*)d_out;

  char* ws = (char*)d_ws;
  size_t off = 0;
  auto give = [&](size_t nelem, size_t esz) -> char* {
    char* p = ws + off;
    off += (nelem * esz + 255) & ~(size_t)255;
    return p;
  };

  __bf16* w_e0f_ih = (__bf16*)give(1024 * 256, 2);
  __bf16* w_e0f_hh = (__bf16*)give(1024 * 256, 2);
  __bf16* w_e0b_ih = (__bf16*)give(1024 * 256, 2);
  __bf16* w_e0b_hh = (__bf16*)give(1024 * 256, 2);
  __bf16* w_e1f_ih = (__bf16*)give(1024 * 512, 2);
  __bf16* w_e1f_hh = (__bf16*)give(1024 * 256, 2);
  __bf16* w_e1b_ih = (__bf16*)give(1024 * 512, 2);
  __bf16* w_e1b_hh = (__bf16*)give(1024 * 256, 2);
  __bf16* w_d0_ih  = (__bf16*)give(2048 * 256, 2);
  __bf16* w_d0_hh  = (__bf16*)give(2048 * 512, 2);
  __bf16* w_d1_ih  = (__bf16*)give(2048 * 512, 2);
  __bf16* w_d1_hh  = (__bf16*)give(2048 * 512, 2);
  __bf16* w_lin    = (__bf16*)give((size_t)VOUT * HD, 2);
  __bf16* w_oemb   = (__bf16*)give((size_t)VOUT * E, 2);
  __bf16* xs_bf    = (__bf16*)give((size_t)S * B * E, 2);
  __bf16* xs1_bf   = (__bf16*)give((size_t)S * B * 2 * H, 2);
  __bf16* h0bf[2]  = { (__bf16*)give(B * HD, 2), (__bf16*)give(B * HD, 2) };
  __bf16* h1bf[2]  = { (__bf16*)give(B * HD, 2), (__bf16*)give(B * HD, 2) };
  float*  hdec0    = (float*)give(B * HD, 4);
  float*  cdec0    = (float*)give(B * HD, 4);
  float*  hdec1    = (float*)give(B * HD, 4);
  float*  cdec1    = (float*)give(B * HD, 4);
  unsigned long long* keys = (unsigned long long*)give(B, 8);
  int* tok = (int*)give(B, 4);

  auto cast = [&](const float* s, __bf16* d, int n) {
    int grid = (n + 255) / 256; if (grid > 4096) grid = 4096;
    k_cast<<<grid, 256, 0, stream>>>(s, d, n);
  };

  // weight / embedding casts to bf16
  cast(e0f_ih, w_e0f_ih, 1024 * 256);  cast(e0f_hh, w_e0f_hh, 1024 * 256);
  cast(e0b_ih, w_e0b_ih, 1024 * 256);  cast(e0b_hh, w_e0b_hh, 1024 * 256);
  cast(e1f_ih, w_e1f_ih, 1024 * 512);  cast(e1f_hh, w_e1f_hh, 1024 * 256);
  cast(e1b_ih, w_e1b_ih, 1024 * 512);  cast(e1b_hh, w_e1b_hh, 1024 * 256);
  cast(d0_ih, w_d0_ih, 2048 * 256);    cast(d0_hh, w_d0_hh, 2048 * 512);
  cast(d1_ih, w_d1_ih, 2048 * 512);    cast(d1_hh, w_d1_hh, 2048 * 512);
  cast(linW, w_lin, VOUT * HD);        cast(o_emb, w_oemb, VOUT * E);

  // encoder
  k_embed<<<4096, 256, 0, stream>>>(inputs, in_emb, xs_bf, S * B * E);
  k_enc_scan<<<2, 512, 0, stream>>>(xs_bf, E, xs1_bf,
                                    w_e0f_ih, w_e0f_hh, e0f_b,
                                    w_e0b_ih, w_e0b_hh, e0b_b, hdec0, cdec0);
  k_enc_scan<<<2, 512, 0, stream>>>(xs1_bf, 2 * H, (__bf16*)nullptr,
                                    w_e1f_ih, w_e1f_hh, e1f_b,
                                    w_e1b_ih, w_e1b_hh, e1b_b, hdec1, cdec1);

  // decoder init
  cast(hdec0, h0bf[0], B * HD);
  cast(hdec1, h1bf[0], B * HD);
  k_init_tok<<<1, 32, 0, stream>>>(tok, cls);
  k_first_row<<<(B * VOUT + 255) / 256, 256, 0, stream>>>(out, cls);

  // greedy autoregressive decode
  int cur = 0;
  for (int t = 1; t < TLEN; ++t) {
    const int nxt = cur ^ 1;
    k_clear_keys<<<1, 32, 0, stream>>>(keys);
    k_dec_cell<<<4, 256, 0, stream>>>((const __bf16*)nullptr, 1, tok, w_oemb, E,
                                      h0bf[cur], h0bf[nxt], cdec0,
                                      w_d0_ih, w_d0_hh, d0_b);
    k_dec_cell<<<4, 256, 0, stream>>>(h0bf[nxt], 0, tok, (const __bf16*)nullptr, HD,
                                      h1bf[cur], h1bf[nxt], cdec1,
                                      w_d1_ih, w_d1_hh, d1_b);
    k_dec_proj<<<32, 256, 0, stream>>>(h1bf[nxt], w_lin, linb, out, t, keys);
    k_next_tok<<<1, 32, 0, stream>>>(keys, tok);
    cur = nxt;
  }
}
